// AttentionHead_4715874091816
// MI455X (gfx1250) — matrix operand order repaired and checked by reading, not compiled, not run on hardware
//
#include <hip/hip_runtime.h>
#include <hip/hip_fp16.h>

// ---------------------------------------------------------------------------
// Flash-attention head for gfx1250 (MI455X), wave32.
// B=8, T=2048, C=1024, HS=64. fp32 in/out, f16 storage + f32 WMMA accumulate.
// Paths: v_wmma_f32_16x16x32_f16, global_load_async_to_lds_b128 (+ASYNCcnt),
//        LDS transpose staging for P (DScnt), fragment-packed operands.
// ---------------------------------------------------------------------------

typedef _Float16 v16h __attribute__((ext_vector_type(16)));
typedef _Float16 v8h  __attribute__((ext_vector_type(8)));
typedef float    v8f  __attribute__((ext_vector_type(8)));
typedef float    v4f  __attribute__((ext_vector_type(4)));

#define WMMA(a, b, c) \
  __builtin_amdgcn_wmma_f32_16x16x32_f16(false, (a), false, (b), (short)0, (c), false, false)

// async copy: 32 lanes x 16B, immediate offset biases BOTH lds and global addrs
#define ACP(l, g, OFF)                                                        \
  asm volatile("global_load_async_to_lds_b128 %0, %1, off offset:" #OFF       \
               :: "v"(l), "v"(g) : "memory")

static constexpr int Tt = 2048;
static constexpr int Cc = 1024;
static constexpr int HS = 64;

// ---------------------------------------------------------------------------
// Kernel 1: repack W (C x HS fp32) into f16 B-fragment layout.
// frag f = m*128 + kc*4 + nt ; per-lane 16 contiguous halfs.
// B 32x16 layout: lane n<16 -> col n, elems 0..7 -> K=8h+e, elems 8..15 -> K=16+8h+(e-8), h = lane>>4.
// ---------------------------------------------------------------------------
__global__ __launch_bounds__(32) void pack_w_kernel(
    const float* __restrict__ Wq, const float* __restrict__ Wk,
    const float* __restrict__ Wv, _Float16* __restrict__ pw) {
  const int frag = blockIdx.x;  // 0..383
  const int lane = threadIdx.x;
  const int m  = frag >> 7;
  const int kc = (frag >> 2) & 31;
  const int nt = frag & 3;
  const float* W = (m == 0) ? Wq : ((m == 1) ? Wk : Wv);
  const int col   = nt * 16 + (lane & 15);
  const int kbase = kc * 32 + ((lane >> 4) * 8);
  _Float16* dst = pw + ((size_t)frag * 32 + lane) * 16;
#pragma unroll
  for (int e = 0; e < 16; ++e)
    dst[e] = (_Float16)W[(size_t)(kbase + e + (e >= 8 ? 8 : 0)) * HS + col];
}

// ---------------------------------------------------------------------------
// Kernel 2: QKV projection, one wave per 16-row tile of flattened (B*T, C) x.
// Q stored f16 [b][t][h] (pre-scaled by 1/sqrt(C)).
// K stored fragment-packed for the S-GEMM's B operand:
//   frag (b, s16, hc): lane = (h>>4 &1)*16 + (t&15), elem = h&15, h = hc*32 + ...
// V stored fragment-packed for the PV-GEMM's B operand:
//   frag (b, kb32, nt): lane = (t&31)>>4 *16 + (h&15), elem = t&15 within half.
// ---------------------------------------------------------------------------
__global__ __launch_bounds__(256) void qkv_kernel(
    const float* __restrict__ x, const _Float16* __restrict__ pw,
    _Float16* __restrict__ qh, _Float16* __restrict__ kpk,
    _Float16* __restrict__ vpk) {
  const int lane = threadIdx.x & 31;
  const int wave = threadIdx.x >> 5;
  const int tile = blockIdx.x * 8 + wave;  // 0..1023
  const int row0 = tile * 16;              // flat row over B*T
  const int mrow = lane & 15;
  const int hi   = lane >> 4;
  const int koff = hi * 8;

  v8f acc[12];
#pragma unroll
  for (int i = 0; i < 12; ++i)
#pragma unroll
    for (int e = 0; e < 8; ++e) acc[i][e] = 0.0f;

  const float* xrow = x + (size_t)(row0 + mrow) * Cc;
  const v16h*  pwf  = reinterpret_cast<const v16h*>(pw);

  for (int kc = 0; kc < 32; ++kc) {
    const int kb = kc * 32;
    // A fragment (16x32 f16): lane row = lane&15; elems 0..7 -> K=kb+koff+e,
    // elems 8..15 -> K=kb+16+koff+(e-8). Two contiguous 8-float runs.
    const v4f* p0 = reinterpret_cast<const v4f*>(xrow + kb + koff);
    const v4f* p1 = reinterpret_cast<const v4f*>(xrow + kb + 16 + koff);
    const v4f f0 = p0[0], f1 = p0[1], f2 = p1[0], f3 = p1[1];
    v16h a;
#pragma unroll
    for (int e = 0; e < 4; ++e) {
      a[e]      = (_Float16)f0[e];
      a[4 + e]  = (_Float16)f1[e];
      a[8 + e]  = (_Float16)f2[e];
      a[12 + e] = (_Float16)f3[e];
    }
#pragma unroll
    for (int m = 0; m < 3; ++m) {
#pragma unroll
      for (int nt = 0; nt < 4; ++nt) {
        const int f = m * 128 + kc * 4 + nt;
        const v16h bf = pwf[(size_t)f * 32 + lane];
        acc[m * 4 + nt] = WMMA(a, bf, acc[m * 4 + nt]);
      }
    }
  }

  // D layout: elem r of acc -> row (r + hi*8), col N = mrow (per N-tile).
  const float qscale = 0.03125f;  // 1/sqrt(1024)
#pragma unroll
  for (int nt = 0; nt < 4; ++nt) {
#pragma unroll
    for (int r = 0; r < 8; ++r) {
      const int t  = row0 + r + hi * 8;  // flat row
      const int h  = nt * 16 + mrow;
      const int b  = t >> 11;            // / T
      const int tt = t & (Tt - 1);
      // Q plain
      qh[(size_t)t * HS + h] = (_Float16)(acc[0 * 4 + nt][r] * qscale);
      // K fragment-packed: s16 = tt>>4, n = tt&15, hc = nt>>1, laneh = nt&1, e = mrow
      {
        const int s16 = tt >> 4, n = tt & 15;
        const size_t idx =
            ((((size_t)b * 128 + s16) * 2 + (nt >> 1)) * 32 + (nt & 1) * 16 + n) * 16 + mrow;
        kpk[idx] = (_Float16)acc[1 * 4 + nt][r];
      }
      // V fragment-packed: kb = tt>>5, kib = tt&31 -> laneh = kib>>4, e = kib&15
      {
        const int kb2 = tt >> 5, kib = tt & 31;
        const size_t idx =
            ((((size_t)b * 64 + kb2) * 4 + nt) * 32 + (kib >> 4) * 16 + mrow) * 16 + (kib & 15);
        vpk[idx] = (_Float16)acc[2 * 4 + nt][r];
      }
    }
  }
}

// ---------------------------------------------------------------------------
// Kernel 3: causal flash attention. One wave per 16-query tile, 32-key blocks.
// K fragments: contiguous packed global b128 loads.
// V fragments: async-prefetched (double-buffered) into LDS, ds_load_b128 reads.
// P transposed D-layout -> A-layout through LDS.
// ---------------------------------------------------------------------------
__global__ __launch_bounds__(128) void attn_kernel(
    const _Float16* __restrict__ qh, const _Float16* __restrict__ kpk,
    const _Float16* __restrict__ vpk, float* __restrict__ out) {
  __shared__ __align__(16) _Float16 vbuf[4][2][2048];  // 4 waves x double 4KB
  __shared__ __align__(16) _Float16 pbuf[4][16 * 32];  // P transpose, 1KB/wave

  const int lane = threadIdx.x & 31;
  const int wave = threadIdx.x >> 5;
  const int tile = blockIdx.x * 4 + wave;  // 0..1023
  const int b    = tile >> 7;
  const int it   = tile & 127;
  const int t0   = it * 16;
  const int mrow = lane & 15;
  const int hi   = lane >> 4;
  const int koff = hi * 8;

  // Q A-fragments (K dim = h), two chunks of 32.
  v16h aq[2];
  {
    const _Float16* qrow = qh + (size_t)(b * Tt + t0 + mrow) * HS;
#pragma unroll
    for (int hc = 0; hc < 2; ++hc) {
      const v8h lo = *reinterpret_cast<const v8h*>(qrow + hc * 32 + hi * 16);
      const v8h hv = *reinterpret_cast<const v8h*>(qrow + hc * 32 + hi * 16 + 8);
#pragma unroll
      for (int e = 0; e < 8; ++e) { aq[hc][e] = lo[e]; aq[hc][8 + e] = hv[e]; }
    }
  }

  v8f o[4];
#pragma unroll
  for (int i = 0; i < 4; ++i)
#pragma unroll
    for (int e = 0; e < 8; ++e) o[i][e] = 0.0f;

  float mr[8], lr[8];
#pragma unroll
  for (int r = 0; r < 8; ++r) { mr[r] = -__builtin_inff(); lr[r] = 0.0f; }

  const v16h* kf = reinterpret_cast<const v16h*>(kpk);
  _Float16*   pb = pbuf[wave];
  const int   nkb = (t0 + 15) / 32 + 1;

  // Async prefetch of the 4 V fragments (4KB) for key block kbn into buffer kbn&1.
  auto prefetch_v = [&](int kbn) {
    const uint64_t g = (uint64_t)(uintptr_t)(vpk + (size_t)(b * 64 + kbn) * 2048) +
                       (uint32_t)(lane * 16);
    const uint32_t l = (uint32_t)(uintptr_t)(&vbuf[wave][kbn & 1][0]) +
                       (uint32_t)(lane * 16);
    ACP(l, g, 0);    ACP(l, g, 512);  ACP(l, g, 1024); ACP(l, g, 1536);
    ACP(l, g, 2048); ACP(l, g, 2560); ACP(l, g, 3072); ACP(l, g, 3584);
  };

  prefetch_v(0);

  for (int kb = 0; kb < nkb; ++kb) {
    const int scol = kb * 32 + mrow;  // this lane's key column (subtile 0)

    // current V buffer is ready; overlap next block's prefetch with S + softmax
    asm volatile("s_wait_asynccnt 0" ::: "memory");
    if (kb + 1 < nkb) prefetch_v(kb + 1);

    // --- S = Q * K^T : packed K fragments, contiguous 32B per lane ---
    v8f S0, S1;
#pragma unroll
    for (int e = 0; e < 8; ++e) { S0[e] = 0.0f; S1[e] = 0.0f; }
#pragma unroll
    for (int hc = 0; hc < 2; ++hc) {
      const v16h bk0 = kf[((size_t)(b * 128 + kb * 2 + 0) * 2 + hc) * 32 + lane];
      const v16h bk1 = kf[((size_t)(b * 128 + kb * 2 + 1) * 2 + hc) * 32 + lane];
      S0 = WMMA(aq[hc], bk0, S0);
      S1 = WMMA(aq[hc], bk1, S1);
    }

    // --- causal mask + online softmax (row r of D lives in elem r, 16 lanes) ---
    float alpha[8];
#pragma unroll
    for (int r = 0; r < 8; ++r) {
      const int t = t0 + r + hi * 8;
      float s0 = (scol <= t)      ? S0[r] : -__builtin_inff();
      float s1 = (scol + 16 <= t) ? S1[r] : -__builtin_inff();
      float v = fmaxf(s0, s1);
#pragma unroll
      for (int msk = 1; msk < 16; msk <<= 1) v = fmaxf(v, __shfl_xor(v, msk, 32));
      const float mn = fmaxf(mr[r], v);
      alpha[r] = __expf(mr[r] - mn);
      mr[r] = mn;
      const float p0 = __expf(s0 - mn);
      const float p1 = __expf(s1 - mn);
      S0[r] = p0; S1[r] = p1;
      float rs = p0 + p1;
#pragma unroll
      for (int msk = 1; msk < 16; msk <<= 1) rs += __shfl_xor(rs, msk, 32);
      lr[r] = lr[r] * alpha[r] + rs;
    }

#pragma unroll
    for (int nt = 0; nt < 4; ++nt)
#pragma unroll
      for (int r = 0; r < 8; ++r) o[nt][r] *= alpha[r];

    // --- transpose P (D layout) -> A fragment via LDS ---
#pragma unroll
    for (int r = 0; r < 8; ++r) {
      const int row = r + hi * 8;
      pb[row * 32 + mrow]      = (_Float16)S0[r];
      pb[row * 32 + mrow + 16] = (_Float16)S1[r];
    }
    asm volatile("s_wait_dscnt 0" ::: "memory");
    v16h pa;
    {
      const v8h lo = *reinterpret_cast<const v8h*>(pb + mrow * 32 + hi * 16);
      const v8h hv = *reinterpret_cast<const v8h*>(pb + mrow * 32 + hi * 16 + 8);
#pragma unroll
      for (int e = 0; e < 8; ++e) { pa[e] = lo[e]; pa[8 + e] = hv[e]; }
    }
    asm volatile("s_wait_dscnt 0" ::: "memory");

    // --- O += P * V : V fragments from async-staged LDS buffer ---
    const v16h* vf = reinterpret_cast<const v16h*>(&vbuf[wave][kb & 1][0]);
#pragma unroll
    for (int nt = 0; nt < 4; ++nt) {
      const v16h bv = vf[nt * 32 + lane];
      o[nt] = WMMA(pa, bv, o[nt]);
    }
  }

  // --- epilogue: out = O / l, fp32 ---
#pragma unroll
  for (int nt = 0; nt < 4; ++nt) {
#pragma unroll
    for (int r = 0; r < 8; ++r) {
      const int t = t0 + r + hi * 8;
      const int h = nt * 16 + mrow;
      out[(size_t)(b * Tt + t) * HS + h] = o[nt][r] / lr[r];
    }
  }
}

// ---------------------------------------------------------------------------
extern "C" void kernel_launch(void* const* d_in, const int* in_sizes, int n_in,
                              void* d_out, int out_size, void* d_ws, size_t ws_size,
                              hipStream_t stream) {
  (void)in_sizes; (void)n_in; (void)out_size; (void)ws_size;
  const float* x  = (const float*)d_in[0];
  const float* Wq = (const float*)d_in[1];
  const float* Wk = (const float*)d_in[2];
  const float* Wv = (const float*)d_in[3];
  float* out = (float*)d_out;

  char* ws = (char*)d_ws;
  _Float16* qh  = (_Float16*)(ws);                            // 2 MB  [b][t][h] f16
  _Float16* kpk = (_Float16*)(ws + (size_t)2 * 1024 * 1024);  // 2 MB  K fragments
  _Float16* vpk = (_Float16*)(ws + (size_t)4 * 1024 * 1024);  // 2 MB  V fragments
  _Float16* pw  = (_Float16*)(ws + (size_t)6 * 1024 * 1024);  // 384 KB packed W

  pack_w_kernel<<<dim3(3 * 32 * 4), dim3(32), 0, stream>>>(Wq, Wk, Wv, pw);
  qkv_kernel<<<dim3(128), dim3(256), 0, stream>>>(x, pw, qh, kpk, vpk);
  attn_kernel<<<dim3(256), dim3(128), 0, stream>>>(qh, kpk, vpk, out);
}
